// TRN_90640989815619
// MI455X (gfx1250) — matrix operand order repaired and checked by reading, not compiled, not run on hardware
//
#include <hip/hip_runtime.h>
#include <hip/hip_bf16.h>

// ---------------------------------------------------------------------------
// Model dims (compile-time; match reference)
// ---------------------------------------------------------------------------
#define D_IN     2048
#define D_HID    1024
#define D_OUT    200
#define D_OUTP   224      // D_OUT padded to multiple of 32 for WMMA K loop
#define D_FUT    8
#define D_SEQ    64
#define BATCH    128
#define D_GATES  (4 * D_HID)
#define D_CAT    (D_IN + D_HID)

typedef __attribute__((ext_vector_type(16))) __bf16 v16bf;
typedef __attribute__((ext_vector_type(8)))  float  v8f;

struct U32x8 { uint4 lo, hi; };   // 32 bytes == v16bf

#define GF_RELU  1
#define GF_ACCUM 2

// LDS B-tile staging: 2 buffers x 64 rows x 80B (32 bf16 data + 16B pad)
#define BROW_BYTES 80
#define BBUF_BYTES (64 * BROW_BYTES)

// ---------------------------------------------------------------------------
// Generic bf16 WMMA GEMM:  C[M,N] = act( A[M,K] @ W[N,K]^T + bias ) (+ C_old)
//  - block = 256 threads = 8 waves; wave w owns rows [16w, 16w+16)
//  - block computes a 128x64 tile; K chunked by 32
//  - B tile (64 rows x 32 K) staged in LDS cooperatively, double-buffered:
//      iter kc: issue global loads for chunk kc+1 (regs), batch-load all B
//      fragments from LDS, 4 WMMAs, ds_store chunk kc+1, one barrier.
//  - A fragment (16x32 bf16, §7.12.2): lane m(=lane&15) holds row m;
//    lanes 0-15 carry K = {0..7,16..23}, lanes 16-31 K = {8..15,24..31}
//  - B fragment (32x16 bf16): lane n(=lane&15) holds column n = W row n;
//    lanes 0-15 carry K = {0..7,16..23}, lanes 16-31 K = {8..15,24..31}
// ---------------------------------------------------------------------------
__global__ __launch_bounds__(256) void k_gemm_bf16(
    const __hip_bfloat16* __restrict__ A, int lda,
    const __hip_bfloat16* __restrict__ W, int ldw,
    const float* __restrict__ bias,
    float* __restrict__ C,
    __hip_bfloat16* __restrict__ Cbf,
    int M, int N, int K, int flags)
{
    __shared__ __align__(16) unsigned char ldsB[2 * BBUF_BYTES];

    const int tid  = threadIdx.x;
    const int lane = tid & 31;
    const int wave = tid >> 5;
    const int mbase = wave * 16;
    const int nblk = blockIdx.x * 64;
    const int lrow = lane & 15;
    const int lhi  = lane >> 4;

    const unsigned short* Ap = reinterpret_cast<const unsigned short*>(A);
    const unsigned short* Wp = reinterpret_cast<const unsigned short*>(W);

    // --- cooperative B fill indexing: 4 threads per row, 16B each ---
    const int frow = tid >> 2;            // 0..63
    const int fq   = tid & 3;             // 0..3  -> bf16 elems [fq*8, fq*8+8)
    int wrow = nblk + frow; if (wrow > N - 1) wrow = N - 1;
    const unsigned short* wsrc_row = Wp + (size_t)wrow * ldw + fq * 8;
    uint4* ldst0 = reinterpret_cast<uint4*>(ldsB + frow * BROW_BYTES + fq * 16);
    uint4* ldst1 = reinterpret_cast<uint4*>(ldsB + BBUF_BYTES + frow * BROW_BYTES + fq * 16);

    // --- accumulators seeded with bias ---
    v8f acc[4];
#pragma unroll
    for (int s = 0; s < 4; ++s) {
        int col = nblk + s * 16 + lrow;
        float bv = 0.f;
        if (bias) bv = bias[col < N ? col : N - 1];
#pragma unroll
        for (int r = 0; r < 8; ++r) acc[s][r] = bv;
    }

    const int arow  = mbase + lrow;
    const unsigned short* arow_p = Ap + (size_t)arow * lda + lhi * 8;
    const int nchunks = K >> 5;

    // --- prologue: fill buffer 0 with chunk 0, prefetch A chunk 0 ---
    {
        uint4 g = *reinterpret_cast<const uint4*>(wsrc_row);   // k0 = 0
        *ldst0 = g;
    }
    U32x8 ua;
    {
        const uint4* pa = reinterpret_cast<const uint4*>(arow_p);
        ua.lo = pa[0]; ua.hi = pa[2];
    }
    __syncthreads();

    for (int kc = 0; kc < nchunks; ++kc) {
        const int nxt = kc + 1;
        uint4 gb_n;
        U32x8 ua_n;
        if (nxt < nchunks) {
            // issue next-chunk global loads early (latency hidden under WMMA)
            gb_n = *reinterpret_cast<const uint4*>(wsrc_row + nxt * 32);
            const uint4* pa = reinterpret_cast<const uint4*>(arow_p + nxt * 32);
            ua_n.lo = pa[0]; ua_n.hi = pa[2];
        }

        // batch-load all 4 B fragments from LDS, then run the 4 WMMAs so the
        // scheduler can overlap WMMA with the remaining ds reads
        const unsigned char* buf = ldsB + (kc & 1) * BBUF_BYTES;
        U32x8 ub[4];
#pragma unroll
        for (int s = 0; s < 4; ++s) {
            const uint4* pb = reinterpret_cast<const uint4*>(
                buf + (s * 16 + lrow) * BROW_BYTES + lhi * 16);
            ub[s].lo = pb[0];
            ub[s].hi = pb[2];
        }
        v16bf af = __builtin_bit_cast(v16bf, ua);
#pragma unroll
        for (int s = 0; s < 4; ++s) {
            v16bf bfr = __builtin_bit_cast(v16bf, ub[s]);
            acc[s] = __builtin_amdgcn_wmma_f32_16x16x32_bf16(
                false, af, false, bfr, (short)0, acc[s], false, false);
        }

        if (nxt < nchunks) {
            // stage next chunk into the other buffer (last read at iter kc-1,
            // protected by the barrier at the end of iter kc-1)
            *((nxt & 1) ? ldst1 : ldst0) = gb_n;
            ua = ua_n;
        }
        __syncthreads();
    }

    // epilogue: C/D layout — VGPR r, lane l: row = r + 8*(l>>4), col = l&15
#pragma unroll
    for (int s = 0; s < 4; ++s) {
        int col = nblk + s * 16 + lrow;
        if (col >= N) continue;
#pragma unroll
        for (int r = 0; r < 8; ++r) {
            int row = mbase + r + lhi * 8;
            float v = acc[s][r];
            if (flags & GF_RELU)  v = fmaxf(v, 0.f);
            size_t idx = (size_t)row * N + col;
            if (flags & GF_ACCUM) v += C[idx];
            C[idx] = v;
            if (Cbf) Cbf[idx] = __float2bfloat16(v);
        }
    }
}

// ---------------------------------------------------------------------------
// fp32 -> bf16 conversion with optional column padding (zeros past scols)
// ---------------------------------------------------------------------------
__global__ void k_to_bf16(const float* __restrict__ src, __hip_bfloat16* __restrict__ dst,
                          int rows, int scols, int dcols)
{
    int i = blockIdx.x * blockDim.x + threadIdx.x;
    if (i >= rows * dcols) return;
    int r = i / dcols, c = i % dcols;
    float v = (c < scols) ? src[(size_t)r * scols + c] : 0.f;
    dst[i] = __float2bfloat16(v);
}

// ---------------------------------------------------------------------------
// LSTM gate nonlinearity (torch order i,f,g,o): c' = sig(f)c + sig(i)tanh(g),
// h' = sig(o)tanh(c').  Writes c' (fp32), h' (bf16), optional c' (bf16).
// ---------------------------------------------------------------------------
__global__ void k_lstm_pt(const float* __restrict__ gates,
                          const float* __restrict__ cprev,
                          float* __restrict__ cout,
                          __hip_bfloat16* __restrict__ hbf,
                          __hip_bfloat16* __restrict__ cbf)
{
    int i = blockIdx.x * blockDim.x + threadIdx.x;
    if (i >= BATCH * D_HID) return;
    int b = i / D_HID, j = i % D_HID;
    const float* g = gates + (size_t)b * D_GATES;
    float gi = g[j], gf = g[j + D_HID], gg = g[j + 2 * D_HID], go = g[j + 3 * D_HID];
    float si = 1.f / (1.f + __expf(-gi));
    float sf = 1.f / (1.f + __expf(-gf));
    float so = 1.f / (1.f + __expf(-go));
    float c  = sf * cprev[i] + si * tanhf(gg);
    float h  = so * tanhf(c);
    cout[i] = c;
    hbf[i]  = __float2bfloat16(h);
    if (cbf) cbf[i] = __float2bfloat16(c);
}

// ---------------------------------------------------------------------------
// Row softmax over D_OUT, wave32 per row, bf16 out padded to D_OUTP with zeros
// ---------------------------------------------------------------------------
__global__ void k_softmax(const float* __restrict__ score, __hip_bfloat16* __restrict__ prob)
{
    int lane = threadIdx.x & 31;
    int row  = blockIdx.x * (blockDim.x >> 5) + (threadIdx.x >> 5);
    if (row >= BATCH) return;
    const float* s = score + (size_t)row * D_OUT;
    float m = -3.4e38f;
    for (int c = lane; c < D_OUT; c += 32) m = fmaxf(m, s[c]);
    for (int o = 16; o; o >>= 1) m = fmaxf(m, __shfl_xor(m, o, 32));
    float sum = 0.f;
    for (int c = lane; c < D_OUT; c += 32) sum += __expf(s[c] - m);
    for (int o = 16; o; o >>= 1) sum += __shfl_xor(sum, o, 32);
    float inv = 1.f / sum;
    __hip_bfloat16* p = prob + (size_t)row * D_OUTP;
    for (int c = lane; c < D_OUTP; c += 32)
        p[c] = __float2bfloat16(c < D_OUT ? __expf(s[c] - m) * inv : 0.f);
}

// ---------------------------------------------------------------------------
// xcat = bf16([ x_t | futsum/8 ])   shape [B, D_CAT]
// ---------------------------------------------------------------------------
__global__ void k_concat(const float* __restrict__ xt, const float* __restrict__ futsum,
                         __hip_bfloat16* __restrict__ xcat)
{
    int i = blockIdx.x * blockDim.x + threadIdx.x;
    if (i >= BATCH * D_CAT) return;
    int b = i / D_CAT, c = i % D_CAT;
    float v = (c < D_IN) ? xt[(size_t)b * D_IN + c]
                         : futsum[(size_t)b * D_HID + (c - D_IN)] * 0.125f;
    xcat[i] = __float2bfloat16(v);
}

// ---------------------------------------------------------------------------
// Host driver
// ---------------------------------------------------------------------------
extern "C" void kernel_launch(void* const* d_in, const int* in_sizes, int n_in,
                              void* d_out, int out_size, void* d_ws, size_t ws_size,
                              hipStream_t stream)
{
    (void)in_sizes; (void)n_in; (void)out_size;

    const float* x       = (const float*)d_in[0];   // [64,128,2048]
    const float* hx0     = (const float*)d_in[1];   // [128,1024]
    const float* cx0     = (const float*)d_in[2];
    const float* enc_Wih = (const float*)d_in[3];   // [4096,3072]
    const float* enc_Whh = (const float*)d_in[4];   // [4096,1024]
    const float* enc_b   = (const float*)d_in[5];
    const float* dec_Wih = (const float*)d_in[6];   // [4096,1024]
    const float* dec_Whh = (const float*)d_in[7];
    const float* dec_b   = (const float*)d_in[8];
    const float* hx_W    = (const float*)d_in[9];   // [1024,1024]
    const float* hx_b    = (const float*)d_in[10];
    const float* cx_W    = (const float*)d_in[11];
    const float* cx_b    = (const float*)d_in[12];
    const float* fus_W   = (const float*)d_in[13];  // [1024,200]
    const float* fus_b   = (const float*)d_in[14];
    const float* fut_W   = (const float*)d_in[15];  // [1024,1024]
    const float* fut_b   = (const float*)d_in[16];
    const float* cls_W   = (const float*)d_in[17];  // [200,1024]
    const float* cls_b   = (const float*)d_in[18];

    float* act_out = (float*)d_out;                       // [64,128,200]
    float* bar_out = act_out + (size_t)D_SEQ * BATCH * D_OUT; // [64,8,128,200]

    // --- workspace carve-out (bf16 weight copies live in L2 ~50MB) ---
    char* wsb = (char*)d_ws;
    size_t off = 0;
    auto alloc = [&](size_t bytes) -> void* {
        void* p = wsb + off;
        off = (off + bytes + 255) & ~(size_t)255;
        return p;
    };
    typedef __hip_bfloat16 bf;
    bf* encWih_bf = (bf*)alloc((size_t)D_GATES * D_CAT * 2);
    bf* encWhh_bf = (bf*)alloc((size_t)D_GATES * D_HID * 2);
    bf* decWih_bf = (bf*)alloc((size_t)D_GATES * D_HID * 2);
    bf* decWhh_bf = (bf*)alloc((size_t)D_GATES * D_HID * 2);
    bf* hxW_bf    = (bf*)alloc((size_t)D_HID * D_HID * 2);
    bf* cxW_bf    = (bf*)alloc((size_t)D_HID * D_HID * 2);
    bf* futW_bf   = (bf*)alloc((size_t)D_HID * D_HID * 2);
    bf* fusW_bf   = (bf*)alloc((size_t)D_HID * D_OUTP * 2);   // K-padded
    bf* clsW_bf   = (bf*)alloc((size_t)D_OUT * D_HID * 2);
    bf* hx_bf     = (bf*)alloc((size_t)BATCH * D_HID * 2);
    bf* cx_bf     = (bf*)alloc((size_t)BATCH * D_HID * 2);
    float* cx_f   = (float*)alloc((size_t)BATCH * D_HID * 4);
    bf* dh_bf     = (bf*)alloc((size_t)BATCH * D_HID * 2);
    float* dh_f   = (float*)alloc((size_t)BATCH * D_HID * 4);
    float* dc_f   = (float*)alloc((size_t)BATCH * D_HID * 4);
    float* gates  = (float*)alloc((size_t)BATCH * D_GATES * 4);
    float* fus_f  = (float*)alloc((size_t)BATCH * D_HID * 4);
    bf* fus_bf    = (bf*)alloc((size_t)BATCH * D_HID * 2);
    bf* prob_bf   = (bf*)alloc((size_t)BATCH * D_OUTP * 2);
    float* futsum = (float*)alloc((size_t)BATCH * D_HID * 4);
    bf* xcat_bf   = (bf*)alloc((size_t)BATCH * D_CAT * 2);
    if (off > ws_size) return;   // workspace too small: bail deterministically

    auto conv = [&](const float* s, bf* d, int rows, int scols, int dcols) {
        int tot = rows * dcols;
        k_to_bf16<<<(tot + 255) / 256, 256, 0, stream>>>(s, d, rows, scols, dcols);
    };
    auto gemm = [&](const bf* A, int lda, const bf* W, int ldw, const float* bias,
                    float* C, bf* Cbf, int N, int K, int flags) {
        k_gemm_bf16<<<dim3((N + 63) / 64), 256, 0, stream>>>(
            A, lda, W, ldw, bias, C, Cbf, BATCH, N, K, flags);
    };

    // --- one-time (per call) bf16 weight conversion ---
    conv(enc_Wih, encWih_bf, D_GATES, D_CAT, D_CAT);
    conv(enc_Whh, encWhh_bf, D_GATES, D_HID, D_HID);
    conv(dec_Wih, decWih_bf, D_GATES, D_HID, D_HID);
    conv(dec_Whh, decWhh_bf, D_GATES, D_HID, D_HID);
    conv(hx_W, hxW_bf, D_HID, D_HID, D_HID);
    conv(cx_W, cxW_bf, D_HID, D_HID, D_HID);
    conv(fut_W, futW_bf, D_HID, D_HID, D_HID);
    conv(fus_W, fusW_bf, D_HID, D_OUT, D_OUTP);   // zero-pad K 200 -> 224
    conv(cls_W, clsW_bf, D_OUT, D_HID, D_HID);
    conv(hx0, hx_bf, BATCH, D_HID, D_HID);
    conv(cx0, cx_bf, BATCH, D_HID, D_HID);
    hipMemcpyAsync(cx_f, cx0, (size_t)BATCH * D_HID * 4, hipMemcpyDeviceToDevice, stream);

    const int BH = BATCH * D_HID;
    for (int t = 0; t < D_SEQ; ++t) {
        // decoder seed: dh = hx@hxW^T + hx_b ; dc = cx@cxW^T + cx_b
        gemm(hx_bf, D_HID, hxW_bf, D_HID, hx_b, dh_f, dh_bf, D_HID, D_HID, 0);
        gemm(cx_bf, D_HID, cxW_bf, D_HID, cx_b, dc_f, nullptr, D_HID, D_HID, 0);
        hipMemsetAsync(fus_bf, 0, (size_t)BH * 2, stream);
        hipMemsetAsync(futsum, 0, (size_t)BH * 4, stream);

        for (int d = 0; d < D_FUT; ++d) {
            // gates = fus@decWih^T + dh@decWhh^T + dec_b   (two-pass accumulate)
            gemm(fus_bf, D_HID, decWih_bf, D_HID, dec_b, gates, nullptr, D_GATES, D_HID, 0);
            gemm(dh_bf, D_HID, decWhh_bf, D_HID, nullptr, gates, nullptr, D_GATES, D_HID, GF_ACCUM);
            k_lstm_pt<<<(BH + 255) / 256, 256, 0, stream>>>(gates, dc_f, dc_f, dh_bf, nullptr);
            // score -> d_out slice all_act_bar[t][d]
            float* score = bar_out + ((size_t)t * D_FUT + d) * BATCH * D_OUT;
            gemm(dh_bf, D_HID, clsW_bf, D_HID, cls_b, score, nullptr, D_OUT, D_HID, 0);
            k_softmax<<<(BATCH + 7) / 8, 256, 0, stream>>>(score, prob_bf);
            // fus = relu(softmax(score)@fusW^T + fus_b)
            gemm(prob_bf, D_OUTP, fusW_bf, D_OUTP, fus_b, fus_f, fus_bf, D_HID, D_OUTP, GF_RELU);
            // futsum += relu(dh@futW^T + fut_b)
            gemm(dh_bf, D_HID, futW_bf, D_HID, fut_b, futsum, nullptr, D_HID, D_HID,
                 GF_RELU | GF_ACCUM);
        }

        // encoder LSTM on cat(x_t, mean(fut))
        k_concat<<<(BATCH * D_CAT + 255) / 256, 256, 0, stream>>>(
            x + (size_t)t * BATCH * D_IN, futsum, xcat_bf);
        gemm(xcat_bf, D_CAT, encWih_bf, D_CAT, enc_b, gates, nullptr, D_GATES, D_CAT, 0);
        gemm(hx_bf, D_HID, encWhh_bf, D_HID, nullptr, gates, nullptr, D_GATES, D_HID, GF_ACCUM);
        k_lstm_pt<<<(BH + 255) / 256, 256, 0, stream>>>(gates, cx_f, cx_f, hx_bf, cx_bf);
        // act = hx@clsW^T + cls_b -> d_out slice all_act[t]
        gemm(hx_bf, D_HID, clsW_bf, D_HID, cls_b,
             act_out + (size_t)t * BATCH * D_OUT, nullptr, D_OUT, D_HID, 0);
    }
}
